// SVD_Encoder_28570122453324
// MI455X (gfx1250) — matrix operand order repaired and checked by reading, not compiled, not run on hardware
//
#include <hip/hip_runtime.h>

// ---------------------------------------------------------------------------
// CDNA5 (gfx1250) WMMA types
// ---------------------------------------------------------------------------
typedef __attribute__((ext_vector_type(16))) __bf16 v16bf;
typedef __attribute__((ext_vector_type(8)))  __bf16 v8bf;
typedef __attribute__((ext_vector_type(4)))  __bf16 v4bf;
typedef __attribute__((ext_vector_type(8)))  float  v8f;

union ABfrag { v16bf v; v8bf h[2]; };

__device__ __forceinline__ __bf16 to_bf(float f) { return (__bf16)f; }

__device__ __forceinline__ v8f wmma_bf16(const v16bf& a, const v16bf& b, const v8f& c) {
    return __builtin_amdgcn_wmma_f32_16x16x32_bf16(false, a, false, b, (short)0, c, false, false);
}

// ---------------------------------------------------------------------------
// Elementwise converters
// ---------------------------------------------------------------------------
// Vectorized: 4 floats -> 4 bf16 per thread (n must be a multiple of 4).
__global__ void cvt_f32_to_bf16_v4(const float* __restrict__ src, __bf16* __restrict__ dst, int n4) {
    int i = blockIdx.x * 256 + threadIdx.x;
    if (i < n4) {
        float4 f = ((const float4*)src)[i];
        v4bf o = { to_bf(f.x), to_bf(f.y), to_bf(f.z), to_bf(f.w) };
        ((v4bf*)dst)[i] = o;
    }
}

// dst[g,k] = bf16(wa[g,k] * wb[k])   (wb is [K,1] -> wb[k])
__global__ void scale_cvt_bf16(const float* __restrict__ wa, const float* __restrict__ wb,
                               __bf16* __restrict__ dst, int n, int K) {
    int i = blockIdx.x * 256 + threadIdx.x;
    if (i < n) dst[i] = to_bf(wa[i] * wb[i % K]);
}

// ---------------------------------------------------------------------------
// NT bf16 WMMA GEMM:  C[M,N] = A[M,K] @ B[N,K]^T (+ bias[N])
// Block = 256 threads = 8 waves (4 M-waves x 2 N-waves). Wave tile = 16x64.
// Workgroup tile = 64(M) x 128(N). Requires M%64==0, N%128==0, K%32==0.
// Fragments (16-bit A 16x32; B 32x16; f32 C 16x16); A/B use contiguous k = 16*(lane/16) + s.
// ---------------------------------------------------------------------------
template <bool OUT_BF16>
__global__ void gemm_nt_bf16(const __bf16* __restrict__ A, const __bf16* __restrict__ B,
                             const float* __restrict__ bias,
                             float* __restrict__ Cf, __bf16* __restrict__ Cb,
                             int M, int N, int K) {
    const int lane  = threadIdx.x & 31;
    const int wave  = threadIdx.x >> 5;
    const int wm    = wave & 3;                     // 0..3 (M direction)
    const int wn    = wave >> 2;                    // 0..1 (N direction)
    const int m0    = (blockIdx.y * 4 + wm) * 16;
    const int n0b   = blockIdx.x * 128 + wn * 64;

    const int arow  = m0 + (lane & 15);             // A row this lane feeds
    const int koffA = (lane >> 4) * 16;             // lanes 0-15: K 0..15 ; 16-31: K 16..31
    const int khB   = (lane >> 4) * 16;             // lanes 0-15: K 0..15 ; 16-31: K 16..31
    const int ncol  = lane & 15;

    v8f acc[4] = {};                                // 4 N-tiles of 16x16 f32

    const __bf16* Arow = A + (size_t)arow * K;
    for (int k0 = 0; k0 < K; k0 += 32) {
        if (k0 + 64 < K) __builtin_prefetch(Arow + k0 + 64, 0, 0);  // global_prefetch_b8
        ABfrag a;
        a.h[0] = *(const v8bf*)(Arow + k0 + koffA);
        a.h[1] = *(const v8bf*)(Arow + k0 + koffA + 8);
#pragma unroll
        for (int t = 0; t < 4; ++t) {
            const int n = n0b + t * 16 + ncol;
            v16bf b = *(const v16bf*)(B + (size_t)n * K + k0 + khB);
            acc[t] = wmma_bf16(a.v, b, acc[t]);
        }
    }

    // Store: lane element v -> (M = m0 + 8*(lane>>4) + v, N = n0b + t*16 + (lane&15))
    const int mrow0 = m0 + (lane >> 4) * 8;
#pragma unroll
    for (int t = 0; t < 4; ++t) {
        const int n = n0b + t * 16 + ncol;
        const float bv = bias ? bias[n] : 0.0f;
#pragma unroll
        for (int v = 0; v < 8; ++v) {
            const float val = acc[t][v] + bv;
            if constexpr (OUT_BF16) Cb[(size_t)(mrow0 + v) * N + n] = to_bf(val);
            else                    Cf[(size_t)(mrow0 + v) * N + n] = val;
        }
    }
}

// ---------------------------------------------------------------------------
// GRU recurrence (persistent): 8 blocks x 16 batch rows, 256 threads = 8 waves.
// Per step: gh[16,1536] = h_bf[16,512] @ whh_bf[1536,512]^T + bhh  (WMMA, LDS h),
// then fused gate elementwise + h update in LDS. out/hT written to d_out.
//
// GEMM is tile-major: each wave owns 12 of the 96 N-tiles and runs the full
// K=512 loop per tile-PAIR (two independent accumulator chains). Only ~48 live
// VGPRs -> high occupancy to hide L2 latency of the 1.5MB whh stream; the
// unrolled k-loop lets the scheduler hoist loads over WMMAs instead of
// draining loadcnt to 0 per WMMA (round-1 bottleneck).
//
// Dynamic LDS: h_f32 32KB + h_bf16 16KB + gh 96KB = 144KB (CDNA5: 320KB/WG).
// ---------------------------------------------------------------------------
__global__ void gru_scan_wmma(const float* __restrict__ gi_rz,   // [8192,1024] (i_r|i_i)
                              const float* __restrict__ i_n_,    // [8192, 512]
                              const float* __restrict__ gf,      // [8192,1024] (f_r|f_i)
                              const __bf16* __restrict__ whh_bf, // [1536, 512] (NT)
                              const float* __restrict__ bhh,     // [1536]
                              float* __restrict__ out,           // [128,64,512]
                              float* __restrict__ hT) {          // [128,512]
    extern __shared__ char smem[];
    float*  h_f = (float*)smem;                          // [16][512]
    __bf16* h_b = (__bf16*)(smem + 16 * 512 * 4);        // [16][512]
    float*  gh  = (float*)(smem + 16 * 512 * 4 + 16 * 512 * 2); // [16][1536]

    const int tid  = threadIdx.x;
    const int lane = tid & 31;
    const int wave = tid >> 5;          // 0..7
    const int br0  = blockIdx.x * 16;   // first batch row of this WG

    for (int i = tid; i < 16 * 512; i += 256) { h_f[i] = 0.0f; h_b[i] = to_bf(0.0f); }
    __syncthreads();

    const int arow  = lane & 15;
    const int koffA = (lane >> 4) * 16;
    const int khB   = (lane >> 4) * 16;
    const int ncol  = lane & 15;
    const int mrow0 = (lane >> 4) * 8;
    const __bf16* hrow = h_b + arow * 512;

    for (int s = 0; s < 64; ++s) {
        // ---- gh = h @ whh^T : 96 N-tiles; wave owns tiles (t*8 + wave), t=0..11,
        //      processed as 6 pairs of independent accumulator chains. ----
#pragma unroll 1
        for (int tp = 0; tp < 6; ++tp) {
            const int n0 = ((tp * 2 + 0) * 8 + wave) * 16 + ncol;
            const int n1 = ((tp * 2 + 1) * 8 + wave) * 16 + ncol;
            const __bf16* B0 = whh_bf + (size_t)n0 * 512 + khB;
            const __bf16* B1 = whh_bf + (size_t)n1 * 512 + khB;
            if (tp + 1 < 6) {   // warm next pair's rows (global_prefetch_b8)
                __builtin_prefetch(whh_bf + (size_t)(n0 + 256) * 512 + khB, 0, 0);
                __builtin_prefetch(whh_bf + (size_t)(n1 + 256) * 512 + khB, 0, 0);
            }
            v8f acc0 = {}, acc1 = {};
#pragma unroll
            for (int k0 = 0; k0 < 512; k0 += 32) {
                ABfrag a;
                a.h[0] = *(const v8bf*)(hrow + k0 + koffA);
                a.h[1] = *(const v8bf*)(hrow + k0 + koffA + 8);
                v16bf b0 = *(const v16bf*)(B0 + k0);
                v16bf b1 = *(const v16bf*)(B1 + k0);
                acc0 = wmma_bf16(a.v, b0, acc0);
                acc1 = wmma_bf16(a.v, b1, acc1);
            }
            const float bv0 = bhh[n0];
            const float bv1 = bhh[n1];
#pragma unroll
            for (int v = 0; v < 8; ++v) {
                gh[(mrow0 + v) * 1536 + n0] = acc0[v] + bv0;
                gh[(mrow0 + v) * 1536 + n1] = acc1[v] + bv1;
            }
        }
        __syncthreads();

        // ---- fused gates + state update ----
        for (int i = tid; i < 16 * 512; i += 256) {
            const int m = i >> 9, j = i & 511;
            const size_t rg = (size_t)(br0 + m) * 64 + s;     // flattened [b,s]
            const float ir  = gi_rz[rg * 1024 + j];
            const float ii  = gi_rz[rg * 1024 + 512 + j];
            const float inn = i_n_[rg * 512 + j];
            const float fr  = gf[rg * 1024 + j];
            const float fi  = gf[rg * 1024 + 512 + j];
            const float hr  = gh[m * 1536 + j];
            const float hi  = gh[m * 1536 + 512 + j];
            const float hn  = gh[m * 1536 + 1024 + j];
            const float hx  = h_f[i];
            const float r   = 1.0f / (1.0f + __expf(-(ir + hr + fr)));
            const float z   = 1.0f / (1.0f + __expf(-(ii + hi + fi)));
            const float nn  = tanhf(inn + r * hn);
            const float hy  = nn + z * (hx - nn);
            out[rg * 512 + j] = hy;
            h_f[i] = hy;
            h_b[i] = to_bf(hy);
        }
        __syncthreads();
    }

    for (int i = tid; i < 16 * 512; i += 256) {
        const int m = i >> 9, j = i & 511;
        hT[(size_t)(br0 + m) * 512 + j] = h_f[i];
    }
}

// ---------------------------------------------------------------------------
// Host launch
// ---------------------------------------------------------------------------
extern "C" void kernel_launch(void* const* d_in, const int* in_sizes, int n_in,
                              void* d_out, int out_size, void* d_ws, size_t ws_size,
                              hipStream_t stream) {
    (void)in_sizes; (void)n_in; (void)out_size; (void)ws_size;

    constexpr int H = 512, NF = 512, B = 128, S = 64, D0 = 1536, D1 = 2048;
    constexpr int M = B * S; // 8192

    const float* x0  = (const float*)d_in[0];
    const float* x1  = (const float*)d_in[1];
    const float* wa0 = (const float*)d_in[2];
    const float* wb0 = (const float*)d_in[3];
    const float* wc0 = (const float*)d_in[4];
    const float* b0  = (const float*)d_in[5];
    const float* wa1 = (const float*)d_in[6];
    const float* wb1 = (const float*)d_in[7];
    const float* wc1 = (const float*)d_in[8];
    const float* b1  = (const float*)d_in[9];
    const float* iw1 = (const float*)d_in[10];
    const float* ib1 = (const float*)d_in[11];
    const float* whh = (const float*)d_in[12];
    const float* bhh = (const float*)d_in[13];

    float* out = (float*)d_out;              // [B,S,H]
    float* hT  = out + (size_t)B * S * H;    // [B,H]

    // ---- workspace carve (256B aligned) ----
    char* ws = (char*)d_ws;
    auto alloc = [&](size_t bytes) -> char* {
        char* p = ws;
        ws += (bytes + 255) & ~size_t(255);
        return p;
    };
    __bf16* x1_bf   = (__bf16*)alloc((size_t)M * D1 * 2);   // 32 MB
    __bf16* x0_bf   = (__bf16*)alloc((size_t)M * D0 * 2);   // 24 MB
    __bf16* wc1_bf  = (__bf16*)alloc((size_t)NF * D1 * 2);
    __bf16* wc0_bf  = (__bf16*)alloc((size_t)NF * D0 * 2);
    __bf16* iw1_bf  = (__bf16*)alloc((size_t)H * D1 * 2);
    __bf16* wa1s_bf = (__bf16*)alloc((size_t)2 * H * NF * 2);
    __bf16* wa0s_bf = (__bf16*)alloc((size_t)2 * H * NF * 2);
    __bf16* whh_bf  = (__bf16*)alloc((size_t)3 * H * H * 2);
    __bf16* t1_bf   = (__bf16*)alloc((size_t)M * NF * 2);   // 8 MB
    __bf16* t0_bf   = (__bf16*)alloc((size_t)M * NF * 2);   // 8 MB
    float*  gi_rz   = (float*)alloc((size_t)M * 2 * H * 4); // 32 MB
    float*  i_n_    = (float*)alloc((size_t)M * H * 4);     // 16 MB
    float*  gf      = (float*)alloc((size_t)M * 2 * H * 4); // 32 MB

    auto blocks4 = [](long n) { return (unsigned)((n / 4 + 255) / 256); };
    auto blocks  = [](long n) { return (unsigned)((n + 255) / 256); };

    // ---- bf16 conversions + SVD mid-scale fold ----
    cvt_f32_to_bf16_v4<<<blocks4((long)M * D1), 256, 0, stream>>>(x1, x1_bf, M * D1 / 4);
    cvt_f32_to_bf16_v4<<<blocks4((long)M * D0), 256, 0, stream>>>(x0, x0_bf, M * D0 / 4);
    cvt_f32_to_bf16_v4<<<blocks4((long)NF * D1), 256, 0, stream>>>(wc1, wc1_bf, NF * D1 / 4);
    cvt_f32_to_bf16_v4<<<blocks4((long)NF * D0), 256, 0, stream>>>(wc0, wc0_bf, NF * D0 / 4);
    cvt_f32_to_bf16_v4<<<blocks4((long)H * D1), 256, 0, stream>>>(iw1, iw1_bf, H * D1 / 4);
    cvt_f32_to_bf16_v4<<<blocks4((long)3 * H * H), 256, 0, stream>>>(whh, whh_bf, 3 * H * H / 4);
    scale_cvt_bf16<<<blocks((long)2 * H * NF), 256, 0, stream>>>(wa1, wb1, wa1s_bf, 2 * H * NF, NF);
    scale_cvt_bf16<<<blocks((long)2 * H * NF), 256, 0, stream>>>(wa0, wb0, wa0s_bf, 2 * H * NF, NF);

    // ---- factored input projections (all NT WMMA GEMMs) ----
    dim3 blk(256);
    // t1 = x1 @ wc1^T        [8192,512] K=2048 -> bf16
    gemm_nt_bf16<true><<<dim3(NF / 128, M / 64), blk, 0, stream>>>(
        x1_bf, wc1_bf, nullptr, nullptr, t1_bf, M, NF, D1);
    // gi_rz = t1 @ (wa1*wb1)^T + b1   [8192,1024] K=512 -> f32
    gemm_nt_bf16<false><<<dim3(2 * H / 128, M / 64), blk, 0, stream>>>(
        t1_bf, wa1s_bf, b1, gi_rz, nullptr, M, 2 * H, NF);
    // i_n = x1 @ iw1^T + ib1          [8192,512] K=2048 -> f32
    gemm_nt_bf16<false><<<dim3(H / 128, M / 64), blk, 0, stream>>>(
        x1_bf, iw1_bf, ib1, i_n_, nullptr, M, H, D1);
    // t0 = x0 @ wc0^T        [8192,512] K=1536 -> bf16
    gemm_nt_bf16<true><<<dim3(NF / 128, M / 64), blk, 0, stream>>>(
        x0_bf, wc0_bf, nullptr, nullptr, t0_bf, M, NF, D0);
    // gf = t0 @ (wa0*wb0)^T + b0      [8192,1024] K=512 -> f32
    gemm_nt_bf16<false><<<dim3(2 * H / 128, M / 64), blk, 0, stream>>>(
        t0_bf, wa0s_bf, b0, gf, nullptr, M, 2 * H, NF);

    // ---- recurrent scan: 8 persistent WGs, WMMA h@whh^T each step ----
    const size_t lds_bytes = 16 * 512 * 4 + 16 * 512 * 2 + 16 * 1536 * 4; // 144 KB
    gru_scan_wmma<<<dim3(B / 16), blk, lds_bytes, stream>>>(
        gi_rz, i_n_, gf, whh_bf, bhh, out, hT);
}
